// Transformer_52252572123687
// MI455X (gfx1250) — matrix operand order repaired and checked by reading, not compiled, not run on hardware
//
#include <hip/hip_runtime.h>
#include <hip/hip_bf16.h>

// ---------------------------------------------------------------------------
// Transformer encoder (L=6, B=2, N=1024, D=1024, H=16, DH=64, FF=4096)
// f16 WMMA (v_wmma_f32_16x16x32_f16) for all matmuls, f32 accumulation,
// f32 residual stream + LayerNorm. Fused LDS-resident attention (no NxN in
// HBM). Weights converted f32->f16 transposed once per launch into d_ws.
// ---------------------------------------------------------------------------

#define LYR 6
#define BB  2
#define NN  1024
#define DD  1024
#define HH  16
#define DH  64
#define INNER (HH * DH)   // 1024
#define FF  (4 * DD)      // 4096
#define ROWS (BB * NN)    // 2048

typedef __attribute__((ext_vector_type(16))) _Float16 v16h;
typedef __attribute__((ext_vector_type(8)))  _Float16 v8h;
typedef __attribute__((ext_vector_type(8)))  float    v8f;

// ---------------- WMMA helpers (layouts per CDNA5 ISA 7.12.2) ---------------

__device__ __forceinline__ v8f wmma_f16(v16h a, v16h b, v8f c) {
  // (neg_a, A, neg_b, B, c_mod, C, reuse_a, reuse_b)
  return __builtin_amdgcn_wmma_f32_16x16x32_f16(false, a, false, b, (short)0, c,
                                                false, false);
}

// A-matrix fragment 16x32 f16, row-major source with leading dim ld.
// lane: r = L&15 (row), kbase = (L>>4)*8; VGPR0-3: K=kbase+0..7,
// VGPR4-7: K=kbase+16..23  -> two 16-byte loads.
__device__ __forceinline__ v16h load_frag_a(const _Float16* p0, int ld,
                                            int row0, int k0, int lane) {
  int r  = lane & 15;
  int kb = (lane >> 4) * 8;
  const _Float16* p = p0 + (size_t)(row0 + r) * ld + k0 + kb;
  v8h lo = *(const v8h*)(p);
  v8h hi = *(const v8h*)(p + 16);
  v16h f;
#pragma unroll
  for (int i = 0; i < 8; ++i) { f[i] = lo[i]; f[i + 8] = hi[i]; }
  return f;
}

// B-matrix fragment 32x16 f16 from a transposed [N,K] row-major source:
// lane: c = L&15 (output column), kbase = (L>>4)*8; VGPR0-3: K=kbase+0..7,
// VGPR4-7: K=kbase+16..23 -> two 16-byte loads.
__device__ __forceinline__ v16h load_frag_b(const _Float16* p0, int ld,
                                            int col0, int k0, int lane) {
  int c  = lane & 15;
  int kb = (lane >> 4) * 8;
  const _Float16* p = p0 + (size_t)(col0 + c) * ld + k0 + kb;
  v8h lo = *(const v8h*)(p);
  v8h hi = *(const v8h*)(p + 16);
  v16h f;
#pragma unroll
  for (int i = 0; i < 8; ++i) { f[i] = lo[i]; f[i + 8] = hi[i]; }
  return f;
}

// ---------------- weight transpose + f32->f16 convert -----------------------
// in: f32 [R, C] row-major.  out: f16 [C, R] (out[c,r] = in[r,c]).
__global__ __launch_bounds__(256) void transpose_f16_kernel(
    const float* __restrict__ in, _Float16* __restrict__ out, int R, int C) {
  __shared__ float t[32][33];
  int tx = threadIdx.x, ty = threadIdx.y;        // block (32, 8)
  int c = blockIdx.x * 32 + tx;
#pragma unroll
  for (int i = 0; i < 4; ++i) {
    int r = blockIdx.y * 32 + ty + i * 8;
    t[ty + i * 8][tx] = in[(size_t)r * C + c];
  }
  __syncthreads();
  int rr = blockIdx.y * 32 + tx;
#pragma unroll
  for (int i = 0; i < 4; ++i) {
    int cc = blockIdx.x * 32 + ty + i * 8;
    out[(size_t)cc * R + rr] = (_Float16)t[tx][ty + i * 8];
  }
}

// ---------------- LayerNorm (f32 in -> f16 out) -----------------------------
__global__ __launch_bounds__(256) void ln_kernel(
    const float* __restrict__ x, const float* __restrict__ g,
    const float* __restrict__ b, _Float16* __restrict__ h) {
  __shared__ float s1[256], s2[256];
  int row = blockIdx.x, tid = threadIdx.x;
  const float4* xr = (const float4*)(x + (size_t)row * DD);
  float4 v = xr[tid];
  s1[tid] = v.x + v.y + v.z + v.w;
  s2[tid] = v.x * v.x + v.y * v.y + v.z * v.z + v.w * v.w;
  __syncthreads();
  for (int off = 128; off > 0; off >>= 1) {
    if (tid < off) { s1[tid] += s1[tid + off]; s2[tid] += s2[tid + off]; }
    __syncthreads();
  }
  float mu   = s1[0] * (1.0f / DD);
  float var  = s2[0] * (1.0f / DD) - mu * mu;
  float rstd = rsqrtf(var + 1e-5f);
  const float* vp = (const float*)&v;
#pragma unroll
  for (int i = 0; i < 4; ++i) {
    int c = tid * 4 + i;
    h[(size_t)row * DD + c] = (_Float16)((vp[i] - mu) * rstd * g[c] + b[c]);
  }
}

// ---------------- generic WMMA GEMM -----------------------------------------
// C[M,N] = A[M,K](f16) * Bt[N,K]^T(f16), f32 accumulate.
// mode 0: store f16 (no bias)            -> Cf16
// mode 1: +bias, exact GELU, store f16   -> Cf16
// mode 2: +bias, residual add (f32 RMW)  -> Cres
// Block: 256 threads = 8 waves; block tile 128x64; wave tile 32x32.
__global__ __launch_bounds__(256) void gemm_kernel(
    const _Float16* __restrict__ A, const _Float16* __restrict__ Bt,
    const float* __restrict__ bias, float* __restrict__ Cres,
    _Float16* __restrict__ Cf16, int M, int N, int K, int mode) {
  int tid = threadIdx.x, wave = tid >> 5, lane = tid & 31;
  int wm = wave >> 1, wn = wave & 1;
  int row0 = blockIdx.x * 128 + wm * 32;
  int col0 = blockIdx.y * 64 + wn * 32;
  v8f c00 = {}, c01 = {}, c10 = {}, c11 = {};
  for (int k0 = 0; k0 < K; k0 += 32) {
    v16h a0 = load_frag_a(A, K, row0, k0, lane);
    v16h a1 = load_frag_a(A, K, row0 + 16, k0, lane);
    v16h b0 = load_frag_b(Bt, K, col0, k0, lane);
    v16h b1 = load_frag_b(Bt, K, col0 + 16, k0, lane);
    c00 = wmma_f16(a0, b0, c00);
    c01 = wmma_f16(a0, b1, c01);
    c10 = wmma_f16(a1, b0, c10);
    c11 = wmma_f16(a1, b1, c11);
  }
  int m0 = (lane >> 4) * 8, n = lane & 15;
#pragma unroll
  for (int t = 0; t < 4; ++t) {
    v8f acc = (t == 0) ? c00 : (t == 1) ? c01 : (t == 2) ? c10 : c11;
    int r0 = row0 + ((t >> 1) ? 16 : 0);
    int col = col0 + ((t & 1) ? 16 : 0) + n;
    float bv = (mode >= 1) ? bias[col] : 0.0f;
#pragma unroll
    for (int i = 0; i < 8; ++i) {
      size_t idx = (size_t)(r0 + m0 + i) * N + col;
      float v = acc[i] + bv;
      if (mode == 0) {
        Cf16[idx] = (_Float16)v;
      } else if (mode == 1) {
        v = 0.5f * v * (1.0f + erff(v * 0.70710678118f));   // exact GELU
        Cf16[idx] = (_Float16)v;
      } else {
        Cres[idx] = Cres[idx] + v;                          // residual RMW
      }
    }
  }
}

// ---------------- QKV repack -------------------------------------------------
// qkv f16 [B*N, 3*INNER] -> Qh,Kh [bh, n, d], Vt [bh, d, n]
__global__ __launch_bounds__(256) void repack_kernel(
    const _Float16* __restrict__ qkv, _Float16* __restrict__ Qh,
    _Float16* __restrict__ Kh, _Float16* __restrict__ Vt) {
  size_t idx = (size_t)blockIdx.x * 256 + threadIdx.x;   // < B*N*INNER
  int b   = (int)(idx >> 20);
  int rem = (int)(idx & ((1u << 20) - 1));
  int n   = rem >> 10;
  int hd  = rem & 1023;
  int h = hd >> 6, d = hd & 63;
  const _Float16* p = qkv + ((size_t)(b * NN + n)) * (3 * INNER) + hd;
  int bh = b * HH + h;
  Qh[((size_t)bh * NN + n) * DH + d] = p[0];
  Kh[((size_t)bh * NN + n) * DH + d] = p[INNER];
  Vt[((size_t)bh * DH + d) * NN + n] = p[2 * INNER];
}

// ---------------- fused attention -------------------------------------------
// One block (128 thr = 4 waves) per (bh, 16 query rows). Scores for the full
// 16x1024 row-block live in LDS (f16). Softmax normalization folded into the
// P@V epilogue via saved row sums.
__global__ __launch_bounds__(128) void attn_kernel(
    const _Float16* __restrict__ Qh, const _Float16* __restrict__ Kh,
    const _Float16* __restrict__ Vt, _Float16* __restrict__ obuf) {
  __shared__ __align__(16) _Float16 Ps[16 * NN];   // 32 KB
  __shared__ float red[16 * 8];
  __shared__ float rowmax[16];
  __shared__ float rowsum[16];

  const int qtile = blockIdx.x;          // 0..63
  const int bh    = blockIdx.y;          // 0..31
  const int tid   = threadIdx.x;
  const int wave  = tid >> 5, lane = tid & 31;
  const int qrow0 = qtile * 16;
  const float scale = 0.125f;            // DH^-0.5

  const _Float16* Qb = Qh + (size_t)bh * NN * DH;
  const _Float16* Kb = Kh + (size_t)bh * NN * DH;
  const _Float16* Vb = Vt + (size_t)bh * DH * NN;

  v16h qa0 = load_frag_a(Qb, DH, qrow0, 0, lane);
  v16h qa1 = load_frag_a(Qb, DH, qrow0, 32, lane);

  // Phase 1: S = (Q K^T) * scale -> LDS f16.  Waves own disjoint col tiles.
  for (int ct = wave; ct < NN / 16; ct += 4) {
    int kv0 = ct * 16;
    v16h kb0 = load_frag_b(Kb, DH, kv0, 0, lane);
    v16h kb1 = load_frag_b(Kb, DH, kv0, 32, lane);
    v8f acc = {};
    acc = wmma_f16(qa0, kb0, acc);
    acc = wmma_f16(qa1, kb1, acc);
    int m0 = (lane >> 4) * 8, n = lane & 15;
#pragma unroll
    for (int i = 0; i < 8; ++i)
      Ps[(m0 + i) * NN + kv0 + n] = (_Float16)(acc[i] * scale);
  }
  __syncthreads();

  // Phase 2: row softmax (leave unnormalized exp in Ps, save row sums).
  {
    int r = tid >> 3, j = tid & 7, s = j * 128;
    float lmax = -3.0e38f;
    for (int c = 0; c < 128; ++c)
      lmax = fmaxf(lmax, (float)Ps[r * NN + s + c]);
    red[r * 8 + j] = lmax;
    __syncthreads();
    if (tid < 16) {
      float mx = red[tid * 8];
      for (int jj = 1; jj < 8; ++jj) mx = fmaxf(mx, red[tid * 8 + jj]);
      rowmax[tid] = mx;
    }
    __syncthreads();
    float mx = rowmax[r], lsum = 0.0f;
    for (int c = 0; c < 128; ++c) {
      float e = __expf((float)Ps[r * NN + s + c] - mx);
      Ps[r * NN + s + c] = (_Float16)e;
      lsum += e;
    }
    red[r * 8 + j] = lsum;
    __syncthreads();
    if (tid < 16) {
      float sm = 0.0f;
      for (int jj = 0; jj < 8; ++jj) sm += red[tid * 8 + jj];
      rowsum[tid] = sm;
    }
    __syncthreads();
  }

  // Phase 3: O = P @ V (wave w -> output cols w*16..+15), A-frags from LDS.
  int d0 = wave * 16;
  v8f acc = {};
  for (int kc = 0; kc < NN / 32; ++kc) {
    v16h pa = load_frag_a(Ps, NN, 0, kc * 32, lane);
    v16h vb = load_frag_b(Vb, NN, d0, kc * 32, lane);
    acc = wmma_f16(pa, vb, acc);
  }
  int b = bh >> 4, h = bh & 15;
  int m0 = (lane >> 4) * 8, n = lane & 15;
#pragma unroll
  for (int i = 0; i < 8; ++i) {
    int m = m0 + i;
    float v = acc[i] / rowsum[m];
    obuf[((size_t)(b * NN + qrow0 + m)) * INNER + h * DH + d0 + n] =
        (_Float16)v;
  }
}

// ---------------- host orchestration ----------------------------------------
extern "C" void kernel_launch(void* const* d_in, const int* in_sizes, int n_in,
                              void* d_out, int out_size, void* d_ws,
                              size_t ws_size, hipStream_t stream) {
  const float* x     = (const float*)d_in[0];
  const float* ln1_g = (const float*)d_in[1];
  const float* ln1_b = (const float*)d_in[2];
  const float* wqkv  = (const float*)d_in[3];
  const float* wout  = (const float*)d_in[4];
  const float* bout  = (const float*)d_in[5];
  const float* ln2_g = (const float*)d_in[6];
  const float* ln2_b = (const float*)d_in[7];
  const float* w1    = (const float*)d_in[8];
  const float* b1    = (const float*)d_in[9];
  const float* w2    = (const float*)d_in[10];
  const float* b2    = (const float*)d_in[11];

  char* ws = (char*)d_ws;
  size_t off = 0;
  auto carve = [&](size_t bytes) {
    void* p = ws + off;
    off += (bytes + 255) & ~(size_t)255;
    return p;
  };
  _Float16* wqkvT = (_Float16*)carve((size_t)LYR * 3 * INNER * DD * 2);
  _Float16* woutT = (_Float16*)carve((size_t)LYR * DD * INNER * 2);
  _Float16* w1T   = (_Float16*)carve((size_t)LYR * FF * DD * 2);
  _Float16* w2T   = (_Float16*)carve((size_t)LYR * DD * FF * 2);
  float*    xbuf  = (float*)carve((size_t)ROWS * DD * 4);
  _Float16* hbuf  = (_Float16*)carve((size_t)ROWS * DD * 2);
  _Float16* qkvb  = (_Float16*)carve((size_t)ROWS * 3 * INNER * 2);
  _Float16* Qhb   = (_Float16*)carve((size_t)BB * HH * NN * DH * 2);
  _Float16* Khb   = (_Float16*)carve((size_t)BB * HH * NN * DH * 2);
  _Float16* Vtb   = (_Float16*)carve((size_t)BB * HH * DH * NN * 2);
  _Float16* obuf  = (_Float16*)carve((size_t)ROWS * INNER * 2);
  _Float16* ffb   = (_Float16*)carve((size_t)ROWS * FF * 2);
  (void)ws_size; (void)in_sizes; (void)n_in;

  // Weights -> f16, transposed to [N,K] (recomputed every call; deterministic).
  dim3 tb(32, 8);
  for (int l = 0; l < LYR; ++l) {
    transpose_f16_kernel<<<dim3((3 * INNER) / 32, DD / 32), tb, 0, stream>>>(
        wqkv + (size_t)l * DD * 3 * INNER, wqkvT + (size_t)l * 3 * INNER * DD,
        DD, 3 * INNER);
    transpose_f16_kernel<<<dim3(DD / 32, INNER / 32), tb, 0, stream>>>(
        wout + (size_t)l * INNER * DD, woutT + (size_t)l * DD * INNER,
        INNER, DD);
    transpose_f16_kernel<<<dim3(FF / 32, DD / 32), tb, 0, stream>>>(
        w1 + (size_t)l * DD * FF, w1T + (size_t)l * FF * DD, DD, FF);
    transpose_f16_kernel<<<dim3(DD / 32, FF / 32), tb, 0, stream>>>(
        w2 + (size_t)l * FF * DD, w2T + (size_t)l * DD * FF, FF, DD);
  }

  hipMemcpyAsync(xbuf, x, (size_t)ROWS * DD * 4, hipMemcpyDeviceToDevice,
                 stream);

  for (int l = 0; l < LYR; ++l) {
    // --- attention block ---
    ln_kernel<<<ROWS, 256, 0, stream>>>(xbuf, ln1_g + l * DD, ln1_b + l * DD,
                                        hbuf);
    gemm_kernel<<<dim3(ROWS / 128, (3 * INNER) / 64), 256, 0, stream>>>(
        hbuf, wqkvT + (size_t)l * 3 * INNER * DD, nullptr, nullptr, qkvb,
        ROWS, 3 * INNER, DD, 0);
    repack_kernel<<<(BB * NN * INNER) / 256, 256, 0, stream>>>(qkvb, Qhb, Khb,
                                                               Vtb);
    attn_kernel<<<dim3(NN / 16, BB * HH), 128, 0, stream>>>(Qhb, Khb, Vtb,
                                                            obuf);
    gemm_kernel<<<dim3(ROWS / 128, DD / 64), 256, 0, stream>>>(
        obuf, woutT + (size_t)l * DD * INNER, bout + l * DD, xbuf, nullptr,
        ROWS, DD, INNER, 2);
    // --- FFN block ---
    ln_kernel<<<ROWS, 256, 0, stream>>>(xbuf, ln2_g + l * DD, ln2_b + l * DD,
                                        hbuf);
    gemm_kernel<<<dim3(ROWS / 128, FF / 64), 256, 0, stream>>>(
        hbuf, w1T + (size_t)l * FF * DD, b1 + (size_t)l * FF, nullptr, ffb,
        ROWS, FF, DD, 1);
    gemm_kernel<<<dim3(ROWS / 128, DD / 64), 256, 0, stream>>>(
        ffb, w2T + (size_t)l * DD * FF, b2 + l * DD, xbuf, nullptr,
        ROWS, DD, FF, 2);
  }

  hipMemcpyAsync(d_out, xbuf, (size_t)out_size * sizeof(float),
                 hipMemcpyDeviceToDevice, stream);
}
